// Conv_M_49409303773352
// MI455X (gfx1250) — compile-verified, matrix operand order checked by reading, not run on hardware
//
#include <hip/hip_runtime.h>

typedef __attribute__((ext_vector_type(16))) __bf16 v16bf;
typedef __attribute__((ext_vector_type(8)))  float  v8f;

#define HW      128
#define CIN     64
#define COUT    64
#define FEAT    1152          // 2 * 9 * 64
#define F1OUT   576
#define F2OUT   4096
#define KSTEPS  36            // 1152 / 32
#define W2_TILES 256          // 4096 / 16
#define W1_TILES 36           // 576 / 16
#define FRAG_U16 512          // one WMMA fragment: 32 lanes * 16 bf16
#define MROWS   64            // positions per block (4 row-blocks of 16)

__device__ __forceinline__ float bf2f(unsigned short u) {
  union { unsigned v; float f; } x; x.v = ((unsigned)u) << 16; return x.f;
}
__device__ __forceinline__ unsigned short f2bf(float f) {
  union { float f; unsigned v; } x; x.f = f;
  unsigned r = x.v + 0x7FFFu + ((x.v >> 16) & 1u);   // round-to-nearest-even
  return (unsigned short)(r >> 16);
}
__device__ __forceinline__ int clampi(int v, int lo, int hi) {
  return v < lo ? lo : (v > hi ? hi : v);
}
// u16 index of feat element (row p, k) inside one row-block's A-fragment array.
// bf16 16x32 A layout: lanes 0-15 hold K {0..7, 16..23}, lanes 16-31 hold {8..15, 24..31}
__device__ __forceinline__ int featIdx(int p, int k) {
  int ks = k >> 5, kk = k & 31;
  int lane = p + ((kk & 8) << 1);            // + 16 if kk in {8..15, 24..31}
  int b = (kk & 7) + ((kk >> 4) << 3);       // byte-pair slot within lane's 16 bf16
  return ks * FRAG_U16 + lane * 16 + b;
}
// Load one A fragment from LDS. The empty asm makes the *offset* opaque, so the
// compiler can neither LICM-hoist all 36 fragment loads out of the column loops
// (which caused scratch spills) nor lose the shared-memory provenance of the
// pointer (which turned these into flat_load; with the offset-only opacity they
// stay ds_load_b128 pairs tracked by DScnt alone).
__device__ __forceinline__ v16bf loadA(const unsigned short* base, int ks, int lane) {
  int off = ks * FRAG_U16 + lane * 16;
  asm volatile("" : "+v"(off));
  return *reinterpret_cast<const v16bf*>(&base[off]);
}

// ---------------------------------------------------------------------------
// Prep: pack W1 / W2 (f32 row-major [k][n]) into bf16 WMMA B-fragment order.
// Fragment (tileN, ks) = 512 u16; each lane reads its 32 contiguous bytes.
// ---------------------------------------------------------------------------
__global__ void pack_weights(const float* __restrict__ W1,
                             const float* __restrict__ W2,
                             unsigned short* __restrict__ ws) {
  const int W2N = W2_TILES * KSTEPS * FRAG_U16;   // 4,718,592
  const int W1N = W1_TILES * KSTEPS * FRAG_U16;   //   663,552
  int tid = blockIdx.x * blockDim.x + threadIdx.x;
  if (tid >= W2N + W1N) return;
  int idx; const float* src; unsigned short* dst; int ncols;
  if (tid < W2N) { idx = tid;       src = W2; dst = ws;       ncols = F2OUT; }
  else           { idx = tid - W2N; src = W1; dst = ws + W2N; ncols = F1OUT; }
  int frag = idx / FRAG_U16;
  int w    = idx & (FRAG_U16 - 1);
  int lane = w >> 4, b = w & 15;
  int tN = frag / KSTEPS, ks = frag % KSTEPS;
  int n  = tN * 16 + (lane & 15);
  int kk = (b & 7) + ((b >> 3) << 4) + ((lane >> 4) << 3);
  int k  = ks * 32 + kk;
  dst[idx] = f2bf(src[(size_t)k * ncols + n]);
}

// ---------------------------------------------------------------------------
// Main fused kernel: one block = 64 consecutive positions (half an image row).
// 16 waves = 4 row-blocks x 4 output-channel stripes. Waves sharing a stripe
// issue identical W2-fragment addresses -> WGP$ hits, 4x less L2 traffic.
// ---------------------------------------------------------------------------
__launch_bounds__(512)
__global__ void fused_kernel(const float* __restrict__ x,
                             const float* __restrict__ m,
                             const float* __restrict__ s,
                             const float* __restrict__ b1,
                             const float* __restrict__ b2,
                             const unsigned short* __restrict__ W2p,
                             const unsigned short* __restrict__ W1p,
                             float* __restrict__ out) {
  __shared__ __align__(32) unsigned short featF[4 * KSTEPS * FRAG_U16]; // 144 KB
  __shared__ __align__(32) unsigned short spl[MROWS * F1OUT];           //  72 KB
  __shared__ float yrs[MROWS * CIN], mrs[MROWS * CIN], srs[MROWS * CIN]; // 48 KB

  const int tid  = threadIdx.x;
  const int lane = tid & 31;
  const int wv   = tid >> 5;        // 16 waves
  const int rb   = wv >> 2;         // row-block 0..3 (16 positions each)
  const int qt   = wv & 3;          // quarter / o-stripe 0..3
  const int hi   = lane >> 4;
  const int ln16 = lane & 15;

  const int tile = blockIdx.x;                 // 1024 tiles of 64 positions
  const int w0 = (tile & 1) * 64;              // pos = tile*64 + p64 -> w = w0 + p64
  const int h  = (tile >> 1) & 127;
  const int bb = tile >> 8;

  const unsigned short* featR = &featF[rb * KSTEPS * FRAG_U16];

  // ---- phase 0: zero reducers, build feat tile (bf16, swizzled) + s patches.
  // k-major / p64-minor: 64 adjacent lanes hit 64 consecutive w -> coalesced.
  for (int i = tid; i < MROWS * CIN; i += 512) { yrs[i] = 0.f; mrs[i] = 0.f; srs[i] = 0.f; }
  for (int i = tid; i < MROWS * FEAT; i += 512) {
    int k = i >> 6, p64 = i & 63;
    int kr = (k < F1OUT) ? k : k - F1OUT;
    int c = kr / 9, r = kr - c * 9;
    int hh = clampi(h + r / 3 - 1, 0, 127);
    int ww = clampi(w0 + p64 + (r % 3) - 1, 0, 127);
    size_t off = (((size_t)(bb * CIN + c) * HW) + hh) * HW + ww;
    float v = (k < F1OUT ? x : m)[off];
    featF[(p64 >> 4) * KSTEPS * FRAG_U16 + featIdx(p64 & 15, k)] = f2bf(v);
    if (k < F1OUT) spl[p64 * F1OUT + k] = f2bf(s[off]);
  }
  __syncthreads();

  // ---- phase 1: w1 = feat @ W1 + b1 via WMMA, fused yr/mr/sr LDS-atomic
  // reduction.  Wave (rb, qt) covers col tiles t1 = qt*9 .. qt*9+8 for its rows.
  {
    const unsigned short* wp0 = W1p + (size_t)(qt * 9) * KSTEPS * FRAG_U16 + lane * 16;
    v16bf bcur = *reinterpret_cast<const v16bf*>(wp0);
    for (int i1 = 0; i1 < 9; ++i1) {
      int t1 = qt * 9 + i1;
      const unsigned short* wp = wp0 + (size_t)i1 * KSTEPS * FRAG_U16;
      float bv = b1[t1 * 16 + ln16];
      v8f acc = {bv, bv, bv, bv, bv, bv, bv, bv};
      for (int ks = 0; ks < KSTEPS - 1; ++ks) {
        v16bf bnxt = *reinterpret_cast<const v16bf*>(wp + (ks + 1) * FRAG_U16);
        v16bf a = loadA(featR, ks, lane);
        acc = __builtin_amdgcn_wmma_f32_16x16x32_bf16(false, a, false, bcur,
                                                      (short)0, acc, false, false);
        bcur = bnxt;
      }
      const unsigned short* wpn = (i1 + 1 < 9) ? wp + (size_t)KSTEPS * FRAG_U16 : wp0;
      v16bf bnxt = *reinterpret_cast<const v16bf*>(wpn);
      {
        v16bf a = loadA(featR, KSTEPS - 1, lane);
        acc = __builtin_amdgcn_wmma_f32_16x16x32_bf16(false, a, false, bcur,
                                                      (short)0, acc, false, false);
      }
      int col = t1 * 16 + ln16;
      int c = col / 9;
      #pragma unroll
      for (int j = 0; j < 8; ++j) {
        int p64 = rb * 16 + j + (hi << 3);
        float w1v = acc[j];
        float aw  = __builtin_fabsf(w1v);
        float ypv = bf2f(featR[featIdx((p64 & 15), col)]);
        float mpv = bf2f(featR[featIdx((p64 & 15), col + F1OUT)]);
        float spv = bf2f(spl[p64 * F1OUT + col]);
        atomicAdd(&yrs[p64 * CIN + c], ypv * w1v);
        atomicAdd(&mrs[p64 * CIN + c], mpv * aw);
        atomicAdd(&srs[p64 * CIN + c], spv * aw);
      }
      bcur = bnxt;
    }
  }
  __syncthreads();

  // ---- phase 2: w2 = feat @ W2 + b2 (never materialized), fused contraction.
  // Wave (rb, qt): rows rb*16..rb*16+15, output channels qt*16..qt*16+15.
  // c processed in groups of 4 (A reuse x4); B fragments double-buffered.
  const int on = qt * 16 + ln16;
  float yac[8], mac[8], sac[8];
  #pragma unroll
  for (int j = 0; j < 8; ++j) { yac[j] = 0.f; mac[j] = 0.f; sac[j] = 0.f; }

  // fragment (cg, cc, ks) lives at wbase + ((cg*16 + cc*4)*KSTEPS + ks)*FRAG_U16
  const unsigned short* wbase = W2p + (size_t)qt * KSTEPS * FRAG_U16 + lane * 16;
  v16bf bcur[4];
  #pragma unroll
  for (int cc = 0; cc < 4; ++cc)
    bcur[cc] = *reinterpret_cast<const v16bf*>(
        wbase + (size_t)(cc * 4) * KSTEPS * FRAG_U16);

  for (int cg = 0; cg < CIN / 4; ++cg) {
    const unsigned short* wp = wbase + (size_t)(cg * 16) * KSTEPS * FRAG_U16;
    v8f acc[4];
    #pragma unroll
    for (int cc = 0; cc < 4; ++cc) {
      float bv = b2[(cg * 4 + cc) * COUT + on];
      acc[cc] = (v8f){bv, bv, bv, bv, bv, bv, bv, bv};
    }
    for (int ks = 0; ks < KSTEPS - 1; ++ks) {
      v16bf bnxt[4];
      #pragma unroll
      for (int cc = 0; cc < 4; ++cc)
        bnxt[cc] = *reinterpret_cast<const v16bf*>(
            wp + ((size_t)(cc * 4) * KSTEPS + ks + 1) * FRAG_U16);
      v16bf a = loadA(featR, ks, lane);
      #pragma unroll
      for (int cc = 0; cc < 4; ++cc)
        acc[cc] = __builtin_amdgcn_wmma_f32_16x16x32_bf16(false, a, false, bcur[cc],
                                                          (short)0, acc[cc],
                                                          false, false);
      #pragma unroll
      for (int cc = 0; cc < 4; ++cc) bcur[cc] = bnxt[cc];
    }
    // last k-step; prefetch next c-group's first fragments under the epilogue
    {
      const unsigned short* wpn =
          (cg + 1 < CIN / 4) ? wp + (size_t)16 * KSTEPS * FRAG_U16 : wbase;
      v16bf bnxt[4];
      #pragma unroll
      for (int cc = 0; cc < 4; ++cc)
        bnxt[cc] = *reinterpret_cast<const v16bf*>(
            wpn + (size_t)(cc * 4) * KSTEPS * FRAG_U16);
      v16bf a = loadA(featR, KSTEPS - 1, lane);
      #pragma unroll
      for (int cc = 0; cc < 4; ++cc)
        acc[cc] = __builtin_amdgcn_wmma_f32_16x16x32_bf16(false, a, false, bcur[cc],
                                                          (short)0, acc[cc],
                                                          false, false);
      #pragma unroll
      for (int cc = 0; cc < 4; ++cc) bcur[cc] = bnxt[cc];
    }
    #pragma unroll
    for (int cc = 0; cc < 4; ++cc) {
      int c = cg * 4 + cc;
      #pragma unroll
      for (int j = 0; j < 8; ++j) {
        int p64 = rb * 16 + j + (hi << 3);
        float w2v = acc[cc][j];
        float aw  = __builtin_fabsf(w2v);
        yac[j] += yrs[p64 * CIN + c] * w2v;
        mac[j] += mrs[p64 * CIN + c] * aw;
        sac[j] += srs[p64 * CIN + c] * aw;
      }
    }
  }

  // ---- store y, m_y/s_y, ones  (outputs concatenated, each [4,64,128,128])
  #pragma unroll
  for (int j = 0; j < 8; ++j) {
    int p64 = rb * 16 + j + (hi << 3);
    size_t off = (((size_t)(bb * COUT + on) * HW) + h) * HW + (w0 + p64);
    out[off]           = yac[j];
    out[4194304 + off] = mac[j] / sac[j];
    out[8388608 + off] = 1.0f;
  }
}

extern "C" void kernel_launch(void* const* d_in, const int* in_sizes, int n_in,
                              void* d_out, int out_size, void* d_ws, size_t ws_size,
                              hipStream_t stream) {
  const float* x  = (const float*)d_in[0];
  const float* m  = (const float*)d_in[1];
  const float* s  = (const float*)d_in[2];
  const float* W1 = (const float*)d_in[3];
  const float* b1 = (const float*)d_in[4];
  const float* W2 = (const float*)d_in[5];
  const float* b2 = (const float*)d_in[6];

  unsigned short* ws = (unsigned short*)d_ws;
  const int W2N = W2_TILES * KSTEPS * FRAG_U16;   // u16 elements
  const int W1N = W1_TILES * KSTEPS * FRAG_U16;
  const int total = W2N + W1N;                    // ~10.8 MB of d_ws

  pack_weights<<<(total + 255) / 256, 256, 0, stream>>>(W1, W2, ws);
  fused_kernel<<<1024, 512, 0, stream>>>(x, m, s, b1, b2,
                                         ws, ws + W2N, (float*)d_out);
}
